// Head_37873021616321
// MI455X (gfx1250) — matrix operand order repaired and checked by reading, compiled, not run on hardware
//
#include <hip/hip_runtime.h>
#include <hip/hip_bf16.h>
#include <stdint.h>

typedef __bf16 bf16_t;
typedef __attribute__((ext_vector_type(16))) __bf16 v16bf;
typedef __attribute__((ext_vector_type(8)))  __bf16 v8bf;
typedef __attribute__((ext_vector_type(4)))  __bf16 v4bf;
typedef __attribute__((ext_vector_type(8)))  float  v8f;
typedef uint32_t u32x4 __attribute__((ext_vector_type(4)));
typedef int      i32x4 __attribute__((ext_vector_type(4)));
typedef int      i32x8 __attribute__((ext_vector_type(8)));

static constexpr int BATCH = 8;
static constexpr int SEQ   = 2048;
static constexpr int EMB   = 1024;
static constexpr int DH    = 128;
static constexpr int NSEG  = 8;
#define SCALE 0.03125f   // 1024^-0.5

#if defined(__has_builtin)
#if __has_builtin(__builtin_amdgcn_sched_group_barrier)
#define SCHED_GROUP(mask, size) __builtin_amdgcn_sched_group_barrier((mask), (size), 0)
#endif
#endif
#ifndef SCHED_GROUP
#define SCHED_GROUP(mask, size)
#endif

// ---------- WMMA helpers (wave32, gfx1250 fragment layouts) ----------

__device__ __forceinline__ v8f zero8() {
  v8f z;
#pragma unroll
  for (int e = 0; e < 8; ++e) z[e] = 0.0f;
  return z;
}

// A-matrix 16x32 bf16: lane(0-15)=row M; elems e -> K = k0+16*half+e
__device__ __forceinline__ v16bf load_a(const bf16_t* __restrict__ base, int ld,
                                        int row0, int k0) {
  int lane = threadIdx.x & 31;
  int l16 = lane & 15, hf = lane >> 4;
  const bf16_t* p = base + (size_t)(row0 + l16) * ld + k0 + hf * 16;
  v8bf lo = *(const v8bf*)(p);
  v8bf hi = *(const v8bf*)(p + 8);
  return __builtin_shufflevector(lo, hi, 0,1,2,3,4,5,6,7,8,9,10,11,12,13,14,15);
}

// B-matrix 32x16 bf16 (K x N), src row-major with rows = N (B = src^T):
// lane(0-15)=col N, elems e -> K = k0 + 16*half + e (one 32B contiguous load)
__device__ __forceinline__ v16bf load_b(const bf16_t* __restrict__ base, int ld,
                                        int n0, int k0) {
  int lane = threadIdx.x & 31;
  int l16 = lane & 15, hf = lane >> 4;
  return *(const v16bf*)(base + (size_t)(n0 + l16) * ld + k0 + hf * 16);
}

__device__ __forceinline__ v8f wmma_bf16(v16bf a, v16bf b, v8f c) {
  return __builtin_amdgcn_wmma_f32_16x16x32_bf16(false, a, false, b, (short)0, c,
                                                 false, false);
}

// ---------- Tensor Data Mover: 2D tile (rows x 32 bf16 cols) global -> LDS ----------
// D# per CDNA5 ISA. All fields wave-uniform.

__device__ __forceinline__ void tdm_load_tile_w(const bf16_t* gsrc, uint32_t lds_off,
                                                uint32_t rows, uint32_t row_stride_elems) {
  uint64_t ga = (uint64_t)(uintptr_t)gsrc;
  u32x4 g0;
  g0[0] = 1u;                                            // count=1, user descriptor
  g0[1] = lds_off;                                       // lds_addr (bytes)
  g0[2] = (uint32_t)(ga & 0xFFFFFFFFu);                  // global_addr[31:0]
  g0[3] = (uint32_t)((ga >> 32) & 0x1FFFFFFu) | (2u << 30); // addr[56:32] | type=2
  const uint32_t dim0 = row_stride_elems;                // tensor width (elems)
  const uint32_t dim1 = rows;                            // tensor height
  const uint32_t t0 = 32, t1 = rows, t2 = 0;             // tile 32 x rows
  const uint64_t s0 = row_stride_elems;                  // dim0 stride (elems)
  i32x8 g1;
  g1[0] = (int)(1u << 16);                               // data_size=1 (2 bytes)
  g1[1] = (int)((dim0 & 0xFFFFu) << 16);                 // dim0[15:0] @ bits 63:48
  g1[2] = (int)((dim0 >> 16) | ((dim1 & 0xFFFFu) << 16));
  g1[3] = (int)((dim1 >> 16) | (t0 << 16));              // tile_dim0 @ 127:112
  g1[4] = (int)(t1 | (t2 << 16));                        // tile_dim1, tile_dim2
  g1[5] = (int)(s0 & 0xFFFFFFFFu);                       // dim0_stride[31:0]
  g1[6] = (int)((uint32_t)(s0 >> 32) & 0xFFFFu);         // dim0_stride[47:32]
  g1[7] = 0;                                             // dim1_stride (unused, 2D)
  i32x4 g2 = {0, 0, 0, 0};
  i32x4 g3 = {0, 0, 0, 0};
#if defined(__clang_major__) && (__clang_major__ >= 23)
  i32x8 g4 = {0, 0, 0, 0, 0, 0, 0, 0};
  __builtin_amdgcn_tensor_load_to_lds(g0, g1, g2, g3, g4, 0);
#else
  __builtin_amdgcn_tensor_load_to_lds(g0, g1, g2, g3, 0);
#endif
}

// ---------- fp32 -> bf16 conversion ----------

__global__ void cvt4_f32_bf16(const float* __restrict__ in, bf16_t* __restrict__ out,
                              int n4) {
  int i = blockIdx.x * blockDim.x + threadIdx.x;
  if (i < n4) {
    float4 f = ((const float4*)in)[i];
    v4bf o;
    o[0] = (bf16_t)f.x; o[1] = (bf16_t)f.y; o[2] = (bf16_t)f.z; o[3] = (bf16_t)f.w;
    ((v4bf*)out)[i] = o;
  }
}

// ---------- projections: Q,K row-major [t][d]; V stored transposed [b][d][t] ----------
// W k-slice (128 x 32) staged to LDS by TDM, double buffered; x A-frag register
// double buffered; DS-read/WMMA interleave forced with sched_group_barrier.

__global__ __launch_bounds__(256) void proj_kernel(const bf16_t* __restrict__ xbf,
                                                   const bf16_t* __restrict__ wall,
                                                   bf16_t* __restrict__ Qbf,
                                                   bf16_t* __restrict__ Kbf,
                                                   bf16_t* __restrict__ Vt) {
  __shared__ __align__(16) bf16_t wtile[2][DH * 32];   // 2 x 8 KB
  int wid  = threadIdx.x >> 5;
  int mtile = blockIdx.x * 8 + wid;          // 16-token tile over B*T
  int mat  = blockIdx.y;                      // 0=Q,1=K,2=V
  const bf16_t* W = wall + (size_t)mat * DH * EMB;
  int row0 = mtile * 16;
  int lane = threadIdx.x & 31;
  int l16 = lane & 15, hf = lane >> 4;
  uint32_t lds0 = (uint32_t)(uintptr_t)(&wtile[0][0]);
  uint32_t lds1 = (uint32_t)(uintptr_t)(&wtile[1][0]);

  v8f acc[8];
#pragma unroll
  for (int n = 0; n < 8; ++n) acc[n] = zero8();

  if (wid == 0) {
    tdm_load_tile_w(W, lds0, DH, EMB);       // k0 = 0 slice
    __builtin_amdgcn_s_wait_tensorcnt(0);
  }
  __syncthreads();

  v16bf a_cur = load_a(xbf, EMB, row0, 0);
  int buf = 0;
  for (int k0 = 0; k0 < EMB; k0 += 32) {
    const bool has_next = (k0 + 32) < EMB;
    if (has_next && wid == 0)
      tdm_load_tile_w(W + (k0 + 32), buf ? lds0 : lds1, DH, EMB);
    v16bf a_next = a_cur;
    if (has_next) a_next = load_a(xbf, EMB, row0, k0 + 32);

    const bf16_t* wl = buf ? &wtile[1][0] : &wtile[0][0];
    v16bf bfr[8];
#pragma unroll
    for (int n = 0; n < 8; ++n)
      bfr[n] = *(const v16bf*)(wl + (n * 16 + l16) * 32 + hf * 16);
#pragma unroll
    for (int n = 0; n < 8; ++n)
      acc[n] = wmma_bf16(a_cur, bfr[n], acc[n]);
    // Desired static schedule: 4 DS reads, then (1 WMMA, 2 DS reads) x6, then
    // 2 WMMAs -> every WMMA has a one-fragment load lead (partial dscnt waits).
    SCHED_GROUP(0x100, 4);
#pragma unroll
    for (int n = 0; n < 6; ++n) {
      SCHED_GROUP(0x008, 1);
      SCHED_GROUP(0x100, 2);
    }
    SCHED_GROUP(0x008, 2);

    if (has_next && wid == 0) __builtin_amdgcn_s_wait_tensorcnt(0);
    __syncthreads();
    a_cur = a_next;
    buf ^= 1;
  }

  if (mat < 2) {
    bf16_t* out = (mat == 0) ? Qbf : Kbf;
#pragma unroll
    for (int n = 0; n < 8; ++n)
#pragma unroll
      for (int r = 0; r < 8; ++r) {
        int t = row0 + r + 8 * hf;
        int d = n * 16 + l16;
        out[(size_t)t * DH + d] = (bf16_t)acc[n][r];
      }
  } else {
#pragma unroll
    for (int n = 0; n < 8; ++n)
#pragma unroll
      for (int r = 0; r < 8; ++r) {
        int t = row0 + r + 8 * hf;
        int d = n * 16 + l16;
        int b = t >> 11, tt = t & (SEQ - 1);
        Vt[((size_t)b * DH + d) * SEQ + tt] = (bf16_t)acc[n][r];
      }
  }
}

// ---------- pass 1: per-column (axis=1 softmax) running max & sum ----------

__global__ __launch_bounds__(256) void attn_colstats(const bf16_t* __restrict__ Qbf,
                                                     const bf16_t* __restrict__ Kbf,
                                                     float* __restrict__ pm,
                                                     float* __restrict__ ps) {
  int wid = threadIdx.x >> 5;
  int gid = blockIdx.x * 8 + wid;            // ((b*128 + jt)*NSEG + seg)
  int seg = gid & (NSEG - 1);
  int jt  = (gid >> 3) & 127;
  int b   = gid >> 10;
  int lane = threadIdx.x & 31;
  int l16 = lane & 15, hf = lane >> 4;
  int j0 = jt * 16;
  int j  = j0 + l16;

  const bf16_t* Qb = Qbf + (size_t)b * SEQ * DH;
  const bf16_t* Kb = Kbf + (size_t)b * SEQ * DH;

  // Q B-fragments are invariant over the whole i-loop: load once.
  v16bf bq[4];
#pragma unroll
  for (int kk = 0; kk < 4; ++kk) bq[kk] = load_b(Qb, DH, j0, kk * 32);

  float m = -1e30f, s = 0.0f;
  int i_begin = seg * (SEQ / NSEG);
  if (i_begin < j0) i_begin = j0;            // causal: only i >= j contributes
  int i_end = (seg + 1) * (SEQ / NSEG);

  // register double-buffer the K A-fragments
  v16bf ak[4];
#pragma unroll
  for (int kk = 0; kk < 4; ++kk) ak[kk] = load_a(Kb, DH, i_begin, kk * 32);

  for (int it = i_begin; it < i_end; it += 16) {
    int itn = (it + 16 < i_end) ? (it + 16) : it;
    v16bf an[4];
#pragma unroll
    for (int kk = 0; kk < 4; ++kk) an[kk] = load_a(Kb, DH, itn, kk * 32);

    v8f c = zero8();
#pragma unroll
    for (int kk = 0; kk < 4; ++kk) c = wmma_bf16(ak[kk], bq[kk], c);

    float v[8];
    float tm = -1e30f;
#pragma unroll
    for (int r = 0; r < 8; ++r) {
      int i = it + r + 8 * hf;
      v[r] = (i >= j) ? c[r] * SCALE : -1e30f;
      tm = fmaxf(tm, v[r]);
    }
    tm = fmaxf(tm, __shfl_xor(tm, 16));
    float mn = fmaxf(m, tm);
    float p = 0.0f;
#pragma unroll
    for (int r = 0; r < 8; ++r) p += __expf(v[r] - mn);
    p += __shfl_xor(p, 16);
    s = s * __expf(m - mn) + p;
    m = mn;
#pragma unroll
    for (int kk = 0; kk < 4; ++kk) ak[kk] = an[kk];
  }
  if (hf == 0) {
    size_t o = (size_t)gid * 16 + l16;
    pm[o] = m;
    ps[o] = s;
  }
}

__global__ void col_reduce(const float* __restrict__ pm, const float* __restrict__ ps,
                           float* __restrict__ colmax, float* __restrict__ colsum) {
  int idx = blockIdx.x * blockDim.x + threadIdx.x;   // 0..B*SEQ-1
  int b = idx >> 11;
  int j = idx & (SEQ - 1);
  int jt = j >> 4, jl = j & 15;
  size_t base = (((size_t)b * 128 + jt) * NSEG) * 16 + jl;
  float M = -1e30f;
  for (int sg = 0; sg < NSEG; ++sg) M = fmaxf(M, pm[base + (size_t)sg * 16]);
  float S = 0.0f;
  for (int sg = 0; sg < NSEG; ++sg)
    S += ps[base + (size_t)sg * 16] * __expf(pm[base + (size_t)sg * 16] - M);
  colmax[idx] = M;
  colsum[idx] = S;
}

// ---------- pass 2: out^T = V^T * P^T, streamed over 32-wide j blocks ----------

__global__ __launch_bounds__(256) void attn_out(const bf16_t* __restrict__ Qbf,
                                                const bf16_t* __restrict__ Kbf,
                                                const bf16_t* __restrict__ Vt,
                                                const float* __restrict__ colmax,
                                                const float* __restrict__ colsum,
                                                float* __restrict__ out) {
  int wid = threadIdx.x >> 5;
  int gid = blockIdx.x * 8 + wid;            // 0..1023
  int itile = gid & 127;
  int b = gid >> 7;
  int i0 = itile * 16;
  int lane = threadIdx.x & 31;
  int l16 = lane & 15, hf = lane >> 4;

  const bf16_t* Qb = Qbf + (size_t)b * SEQ * DH;
  const bf16_t* Kb = Kbf + (size_t)b * SEQ * DH;
  const bf16_t* Vb = Vt  + (size_t)b * DH * SEQ;
  const float* cmb = colmax + (size_t)b * SEQ;
  const float* csb = colsum + (size_t)b * SEQ;

  int i = i0 + l16;                           // lane owns output row i (C' column)

  // K B-fragments are invariant over the j-loop: load once.
  v16bf bk[4];
#pragma unroll
  for (int kk = 0; kk < 4; ++kk) bk[kk] = load_b(Kb, DH, i0, kk * 32);

  v8f acc[8];
#pragma unroll
  for (int n = 0; n < 8; ++n) acc[n] = zero8();

  for (int j0 = 0; j0 < i0 + 16; j0 += 32) {
    // S^T tiles: C'[j,i] = Q_j . K_i, two 16-row j tiles
    v16bf a0k[4], a1k[4];
#pragma unroll
    for (int kk = 0; kk < 4; ++kk) {
      a0k[kk] = load_a(Qb, DH, j0, kk * 32);
      a1k[kk] = load_a(Qb, DH, j0 + 16, kk * 32);
    }
    v8f c0 = zero8(), c1 = zero8();
#pragma unroll
    for (int kk = 0; kk < 4; ++kk) {
      c0 = wmma_bf16(a0k[kk], bk[kk], c0);
      c1 = wmma_bf16(a1k[kk], bk[kk], c1);
    }
    // Repack C' (j over regs+halves) into B-fragment layout (j over 16 elems/lane):
    // lane needs j_local = 16*half + e; one half-swap supplies the missing 8.
    float vals[16];
#pragma unroll
    for (int r = 0; r < 8; ++r) {
      float send = hf ? c0[r] : c1[r];
      float recv = __shfl_xor(send, 16);
      vals[r]     = hf ? recv : c0[r];        // j = j0 + 16*half + r
      vals[8 + r] = hf ? c1[r] : recv;        // j = j0 + 16*half + 8 + r
    }
    int jb = j0 + 16 * hf;
    // 16 consecutive colmax/colsum values per lane, 64B aligned -> float4 loads.
    float cmv[16], csv[16];
    const float4* cm4 = (const float4*)(cmb + jb);
    const float4* cs4 = (const float4*)(csb + jb);
#pragma unroll
    for (int q = 0; q < 4; ++q) {
      float4 a = cm4[q];
      float4 d = cs4[q];
      cmv[q * 4 + 0] = a.x; cmv[q * 4 + 1] = a.y;
      cmv[q * 4 + 2] = a.z; cmv[q * 4 + 3] = a.w;
      csv[q * 4 + 0] = d.x; csv[q * 4 + 1] = d.y;
      csv[q * 4 + 2] = d.z; csv[q * 4 + 3] = d.w;
    }
    v16bf pb;
#pragma unroll
    for (int e = 0; e < 16; ++e) {
      int j = jb + e;
      float p = 0.0f;
      if (j <= i) p = __expf(vals[e] * SCALE - cmv[e]) * (1.0f / csv[e]);
      pb[e] = (bf16_t)p;
    }
    // out^T[d,i] += V^T[d,j] * P^T[j,i]; 2-deep pipelined V^T fragment loads.
    v16bf v_cur = load_a(Vb, SEQ, 0, j0);
#pragma unroll
    for (int n = 0; n < 8; ++n) {
      v16bf v_next = v_cur;
      if (n < 7) v_next = load_a(Vb, SEQ, (n + 1) * 16, j0);
      acc[n] = wmma_bf16(v_cur, pb, acc[n]);
      v_cur = v_next;
    }
  }

#pragma unroll
  for (int n = 0; n < 8; ++n)
#pragma unroll
    for (int r = 0; r < 8; ++r) {
      int d = n * 16 + r + 8 * hf;
      out[(((size_t)b * SEQ) + i) * DH + d] = acc[n][r];
    }
}

// ---------- launch ----------

extern "C" void kernel_launch(void* const* d_in, const int* in_sizes, int n_in,
                              void* d_out, int out_size, void* d_ws, size_t ws_size,
                              hipStream_t stream) {
  const float* x  = (const float*)d_in[0];
  const float* Wq = (const float*)d_in[1];
  const float* Wk = (const float*)d_in[2];
  const float* Wv = (const float*)d_in[3];

  char* w = (char*)d_ws;
  size_t off = 0;
  auto take = [&](size_t bytes) -> char* {
    char* p = w + off;
    off += (bytes + 255) & ~(size_t)255;
    return p;
  };
  bf16_t* xbf  = (bf16_t*)take((size_t)BATCH * SEQ * EMB * 2);   // 32 MB
  bf16_t* wall = (bf16_t*)take((size_t)3 * DH * EMB * 2);        // 768 KB
  bf16_t* Qbf  = (bf16_t*)take((size_t)BATCH * SEQ * DH * 2);    // 4 MB
  bf16_t* Kbf  = (bf16_t*)take((size_t)BATCH * SEQ * DH * 2);
  bf16_t* Vt   = (bf16_t*)take((size_t)BATCH * SEQ * DH * 2);
  float*  pm   = (float*)take((size_t)BATCH * 128 * NSEG * 16 * 4);
  float*  ps   = (float*)take((size_t)BATCH * 128 * NSEG * 16 * 4);
  float*  cmax = (float*)take((size_t)BATCH * SEQ * 4);
  float*  csum = (float*)take((size_t)BATCH * SEQ * 4);

  int nx4 = BATCH * SEQ * EMB / 4;
  cvt4_f32_bf16<<<(nx4 + 255) / 256, 256, 0, stream>>>(x, xbf, nx4);
  int nw4 = DH * EMB / 4;
  cvt4_f32_bf16<<<(nw4 + 255) / 256, 256, 0, stream>>>(Wq, wall + 0 * DH * EMB, nw4);
  cvt4_f32_bf16<<<(nw4 + 255) / 256, 256, 0, stream>>>(Wk, wall + 1 * DH * EMB, nw4);
  cvt4_f32_bf16<<<(nw4 + 255) / 256, 256, 0, stream>>>(Wv, wall + 2 * DH * EMB, nw4);

  proj_kernel<<<dim3(128, 3), 256, 0, stream>>>(xbf, wall, Qbf, Kbf, Vt);
  attn_colstats<<<(BATCH * 128 * NSEG) / 8, 256, 0, stream>>>(Qbf, Kbf, pm, ps);
  col_reduce<<<(BATCH * SEQ) / 256, 256, 0, stream>>>(pm, ps, cmax, csum);
  attn_out<<<(BATCH * 128) / 8, 256, 0, stream>>>(Qbf, Kbf, Vt, cmax, csum,
                                                  (float*)d_out);
}
